// SeparableFCTP_18726057411359
// MI455X (gfx1250) — matrix operand order repaired and checked by reading, not compiled, not run on hardware
//
#include <hip/hip_runtime.h>
#include <math.h>

// ---------------------------------------------------------------------------
// SeparableFCTP for MI455X (gfx1250, wave32). All 4 GEMMs on
// v_wmma_f32_16x16x32_bf16. Roofline: ~92 GFLOP, ~0.71 GB HBM (AI~130) ->
// WMMA makes it bandwidth-limited; the dominant stream is then the *weight*
// B-fragments from L2. This version processes 32 edges per workgroup and
// folds d1's 3 vector components into extra M-tiles, so each B-fragment
// register load feeds 2 WMMAs (lin_w0/rad_w2) or 6 WMMAs (lin_w1):
// weight traffic drops from ~4.9 GB to ~2.0 GB. LDS use is ~124.5 KB
// (dynamic) -> 2 workgroups / 320 KB WGP.
// ---------------------------------------------------------------------------

typedef __attribute__((ext_vector_type(16))) __bf16 v16bf;
typedef __attribute__((ext_vector_type(8)))  float  v8f;

#define EDGES     131072
#define EPB       32          // edges per block (2 M-tiles of 16)
#define NODE_DIM  640
#define OUT_DIM   640

#define INV_SQRT3    0.57735026919f
#define INV_SQRT2    0.70710678119f
#define INV_SQRT_FAN 0.05103103630798288f   // 1/sqrt(384)
// analytic N(0,1) second-moment normalizers (reference derives from RNG draws)
#define C_SILU       1.6790f
#define C_SIGMOID    1.8484f

// packed-weight tiles: 32(K) x 16(N) = 512 bf16 = 1 KB, B-fragment order
#define RW1_TILES 8     // (64/32)*(64/16)
#define RW2_TILES 96    // (64/32)*(768/16)
#define LW0_TILES 288   // (384/32)*(384/16)
#define LW1_TILES 96    // (384/32)*(128/16)
#define RW1_OFF 0
#define RW2_OFF (RW1_TILES*512)
#define LW0_OFF (RW2_OFF + RW2_TILES*512)
#define LW1_OFF (LW0_OFF + LW0_TILES*512)
// ws use: 488 KB + 1 KB slack

// dynamic LDS partition (bytes)
#define SM_HA    0        //  4096 B : h  (2 mt x 2 kt x 512 bf16) swizzled
#define SM_D0    4096     // 24576 B : d0 (2 mt x 12 kt x 512 bf16)
#define SM_D1    28672    // 73728 B : d1 (6 groups = comp*2+mt)
#define SM_GATE  102400   // 16384 B : gates f32 (32 x 128)
#define SM_H     118784   //  8192 B : h f32 (32 x 64)
#define SM_ATTR  126976   //   512 B : edge_attr (32 x 4)
#define SMEM_BYTES 127488

__device__ __forceinline__ v8f wmma_bf16(v16bf a, v16bf b, v8f c) {
  return __builtin_amdgcn_wmma_f32_16x16x32_bf16(false, a, false, b, (short)0, c,
                                                 false, false);
}

// Store scalar (row m in tile, col k) into A-fragment-swizzled bf16 buffer.
// contiguous k order used here (B packed alike): element e = k&15; lane =
// m + 16*((k>>4)&1).
__device__ __forceinline__ void store_a(__bf16* buf, int m, int k, float val) {
  int ktile = k >> 5, kin = k & 31;
  int kh = (kin >> 4) & 1;
  int e  = kin & 15;
  buf[(ktile * 32 + m + (kh << 4)) * 16 + e] = (__bf16)val;
}

// ---------------------------------------------------------------------------
// Pack kernel: f32 (K x N row-major) -> bf16 B-fragment tiles.
// tileIdx = nt*(K/32)+kt ; lane holds 16 contiguous bf16: K-run of 16 at
// column n = nt*16+(lane&15), start k = kt*32+(lane>>4)*16.
// ---------------------------------------------------------------------------
__global__ void pack_kernel(const float* __restrict__ w, __bf16* __restrict__ dst,
                            int K, int N) {
  int t = blockIdx.x * blockDim.x + threadIdx.x;
  int Kt = K >> 5;
  int total = Kt * (N >> 4) * 32;
  if (t >= total) return;
  int tile = t >> 5, lane = t & 31;
  int nt = tile / Kt, kt = tile - nt * Kt;
  int n  = nt * 16 + (lane & 15);
  int kb = kt * 32 + (lane >> 4) * 16;
  __bf16* o = dst + (size_t)t * 16;
#pragma unroll
  for (int e = 0; e < 16; ++e)
    o[e] = (__bf16)w[(size_t)(kb + e) * N + n];
}

// ---------------------------------------------------------------------------
// Main fused kernel: 32 edges per 128-thread (4-wave) workgroup.
// ---------------------------------------------------------------------------
__global__ __launch_bounds__(128) void fctp_main(
    const float* __restrict__ node_input,
    const float* __restrict__ edge_attr,
    const float* __restrict__ edge_scalars,
    const float* __restrict__ rad_ln_g,
    const float* __restrict__ rad_ln_b,
    const float* __restrict__ rad_offset,
    const float* __restrict__ lin_b0,
    const __bf16* __restrict__ pack,
    float* __restrict__ out) {
  extern __shared__ __align__(128) char smem[];
  __bf16* sHA   = (__bf16*)(smem + SM_HA);
  __bf16* sD0   = (__bf16*)(smem + SM_D0);
  __bf16* sD1   = (__bf16*)(smem + SM_D1);
  float*  sGate = (float*)(smem + SM_GATE);
  float*  sH    = (float*)(smem + SM_H);
  float*  sAttr = (float*)(smem + SM_ATTR);

  const int tid   = threadIdx.x;
  const int wave  = tid >> 5;
  const int lane  = tid & 31;
  const int ln    = lane & 15;   // N column (B/C/D frag) or M row (A frag)
  const int khalf = lane >> 4;
  const int e0    = blockIdx.x * EPB;

  // ---- stage 0: stage edge_attr (32x4) -------------------------------------
  sAttr[tid] = edge_attr[(size_t)e0 * 4 + tid];

  // ---- stage 1: h = edge_scalars(32x64) @ rad_w1(64x64); ntile = wave ------
  {
    v8f acc0 = {}, acc1 = {};
#pragma unroll
    for (int kt = 0; kt < 2; ++kt) {
      v16bf b = *(const v16bf*)(pack + RW1_OFF + (size_t)((wave * 2 + kt) * 32 + lane) * 16);
#pragma unroll
      for (int mt = 0; mt < 2; ++mt) {
        const float* p0 = edge_scalars + (size_t)(e0 + mt * 16 + ln) * 64 + kt * 32 + khalf * 16;
        v16bf a;
#pragma unroll
        for (int j = 0; j < 16; ++j) {
          a[j]     = (__bf16)p0[j];
        }
        if (mt == 0) acc0 = wmma_bf16(a, b, acc0);
        else         acc1 = wmma_bf16(a, b, acc1);
      }
    }
#pragma unroll
    for (int r = 0; r < 8; ++r) {
      sH[(0 * 16 + r + 8 * khalf) * 64 + wave * 16 + ln] = acc0[r];
      sH[(1 * 16 + r + 8 * khalf) * 64 + wave * 16 + ln] = acc1[r];
    }
  }
  __syncthreads();

  // ---- stage 2: LayerNorm + SiLU over 64, re-swizzle to bf16 A layout ------
  {
    const int row = tid >> 2;   // 0..31 (4 threads/row, adjacent lanes)
    const int off = tid & 3;    // 16 cols each
    float v[16], s = 0.f, ss = 0.f;
#pragma unroll
    for (int j = 0; j < 16; ++j) {
      float x = sH[row * 64 + off * 16 + j];
      v[j] = x; s += x; ss += x * x;
    }
#pragma unroll
    for (int msk = 1; msk < 4; msk <<= 1) {
      s  += __shfl_xor(s,  msk, 32);
      ss += __shfl_xor(ss, msk, 32);
    }
    float mean = s * (1.f / 64.f);
    float var  = ss * (1.f / 64.f) - mean * mean;
    float inv  = rsqrtf(var + 1e-5f);
    __bf16* dst = sHA + (row >> 4) * 1024;   // mt block
    int m = row & 15;
#pragma unroll
    for (int j = 0; j < 16; ++j) {
      int c = off * 16 + j;
      float x  = (v[j] - mean) * inv * rad_ln_g[c] + rad_ln_b[c];
      store_a(dst, m, c, x / (1.f + __expf(-x)));
    }
  }
  __syncthreads();

  // ---- stage 3: w = h @ rad_w2 (+offset), fused tensor-product epilogue ----
  for (int t = wave; t < 48; t += 4) {          // 48 N-tiles of w, 12 per wave
    v8f acc0 = {}, acc1 = {};
#pragma unroll
    for (int kt = 0; kt < 2; ++kt) {
      v16bf b  = *(const v16bf*)(pack + RW2_OFF + (size_t)((t * 2 + kt) * 32 + lane) * 16);
      v16bf a0 = *(const v16bf*)(sHA + 0 * 1024 + (kt * 32 + lane) * 16);
      v16bf a1 = *(const v16bf*)(sHA + 1 * 1024 + (kt * 32 + lane) * 16);
      acc0 = wmma_bf16(a0, b, acc0);
      acc1 = wmma_bf16(a1, b, acc1);
    }
    int   n    = t * 16 + ln;                   // w-space column, region-uniform
    float woff = rad_offset[n];
#pragma unroll
    for (int mt = 0; mt < 2; ++mt) {
#pragma unroll
      for (int r = 0; r < 8; ++r) {
        int    m  = r + 8 * khalf;
        int    mm = mt * 16 + m;                // edge row within block
        size_t e  = (size_t)(e0 + mm);
        float  w  = (mt == 0 ? acc0[r] : acc1[r]) + woff;
        float s1x = sAttr[mm * 4 + 1], s1y = sAttr[mm * 4 + 2], s1z = sAttr[mm * 4 + 3];
        if (n < 256) {                          // wA -> d0[:, u]
          float x0 = node_input[e * NODE_DIM + n];
          store_a(sD0 + mt * 6144, m, n, x0 * sAttr[mm * 4 + 0] * w);
        } else if (n < 512) {                   // wC -> d1[comp][:, u]
          int   u  = n - 256;
          float x0 = node_input[e * NODE_DIM + u];
          float bs = x0 * w;
          store_a(sD1 + (0 * 2 + mt) * 6144, m, u, bs * s1x);
          store_a(sD1 + (1 * 2 + mt) * 6144, m, u, bs * s1y);
          store_a(sD1 + (2 * 2 + mt) * 6144, m, u, bs * s1z);
        } else if (n < 640) {                   // wB -> d0[:, 256+u]
          int u = n - 512;
          const float* xp = node_input + e * NODE_DIM + 256 + (size_t)u * 3;
          float dot = xp[0] * s1x + xp[1] * s1y + xp[2] * s1z;
          store_a(sD0 + mt * 6144, m, 256 + u, dot * INV_SQRT3 * w);
        } else {                                // wD -> d1[comp][:, 256+u] (cross)
          int u = n - 640;
          const float* xp = node_input + e * NODE_DIM + 256 + (size_t)u * 3;
          float cc = INV_SQRT2 * w;
          store_a(sD1 + (0 * 2 + mt) * 6144, m, 256 + u, (xp[1] * s1z - xp[2] * s1y) * cc);
          store_a(sD1 + (1 * 2 + mt) * 6144, m, 256 + u, (xp[2] * s1x - xp[0] * s1z) * cc);
          store_a(sD1 + (2 * 2 + mt) * 6144, m, 256 + u, (xp[0] * s1y - xp[1] * s1x) * cc);
        }
      }
    }
  }
  __syncthreads();

  // ---- stage 4: out0 = d0 @ lin_w0 ; silu->out, sigmoid->gates -------------
  for (int j = 0; j < 6; ++j) {
    int t = j * 4 + wave;                       // 0..23
    __builtin_prefetch(pack + LW0_OFF + (size_t)(((t + 4) % 24) * 12 * 32) * 16, 0, 1);
    v8f acc0 = {}, acc1 = {};
#pragma unroll
    for (int kt = 0; kt < 12; ++kt) {
      v16bf b  = *(const v16bf*)(pack + LW0_OFF + (size_t)((t * 12 + kt) * 32 + lane) * 16);
      v16bf a0 = *(const v16bf*)(sD0 + 0 * 6144 + (kt * 32 + lane) * 16);
      v16bf a1 = *(const v16bf*)(sD0 + 1 * 6144 + (kt * 32 + lane) * 16);
      acc0 = wmma_bf16(a0, b, acc0);
      acc1 = wmma_bf16(a1, b, acc1);
    }
    int   n    = t * 16 + ln;
    float bias = lin_b0[n];
#pragma unroll
    for (int mt = 0; mt < 2; ++mt) {
#pragma unroll
      for (int r = 0; r < 8; ++r) {
        int   mm = mt * 16 + r + 8 * khalf;
        float o  = (mt == 0 ? acc0[r] : acc1[r]) * INV_SQRT_FAN + bias;
        float sg = 1.f / (1.f + __expf(-o));
        if (n < 256) {
          out[(size_t)(e0 + mm) * OUT_DIM + n] = C_SILU * o * sg;
        } else {
          sGate[mm * 128 + (n - 256)] = C_SIGMOID * sg;
        }
      }
    }
  }
  __syncthreads();

  // ---- stage 5: out1 = d1 @ lin_w1 ; 6 M-groups share each B fragment ------
  for (int nt = wave; nt < 8; nt += 4) {        // 2 N-tiles per wave
    v8f acc[6];
#pragma unroll
    for (int g = 0; g < 6; ++g) acc[g] = (v8f){};
#pragma unroll
    for (int kt = 0; kt < 12; ++kt) {
      v16bf b = *(const v16bf*)(pack + LW1_OFF + (size_t)((nt * 12 + kt) * 32 + lane) * 16);
#pragma unroll
      for (int g = 0; g < 6; ++g) {             // g = comp*2 + mt
        v16bf a = *(const v16bf*)(sD1 + g * 6144 + (kt * 32 + lane) * 16);
        acc[g] = wmma_bf16(a, b, acc[g]);
      }
    }
    int v = nt * 16 + ln;
#pragma unroll
    for (int g = 0; g < 6; ++g) {
      int comp = g >> 1, mt = g & 1;
#pragma unroll
      for (int r = 0; r < 8; ++r) {
        int   mm = mt * 16 + r + 8 * khalf;
        float o  = acc[g][r] * INV_SQRT_FAN * sGate[mm * 128 + v];
        out[(size_t)(e0 + mm) * OUT_DIM + 256 + (size_t)v * 3 + comp] = o;
      }
    }
  }
}

// ---------------------------------------------------------------------------
extern "C" void kernel_launch(void* const* d_in, const int* in_sizes, int n_in,
                              void* d_out, int out_size, void* d_ws, size_t ws_size,
                              hipStream_t stream) {
  (void)in_sizes; (void)n_in; (void)out_size; (void)ws_size;
  const float* node_input   = (const float*)d_in[0];
  const float* edge_attr    = (const float*)d_in[1];
  const float* edge_scalars = (const float*)d_in[2];
  const float* rad_w1       = (const float*)d_in[3];
  const float* rad_ln_g     = (const float*)d_in[4];
  const float* rad_ln_b     = (const float*)d_in[5];
  const float* rad_w2       = (const float*)d_in[6];
  const float* rad_offset   = (const float*)d_in[7];
  const float* lin_w0       = (const float*)d_in[8];
  const float* lin_b0       = (const float*)d_in[9];
  const float* lin_w1       = (const float*)d_in[10];
  float*  out  = (float*)d_out;
  __bf16* pack = (__bf16*)d_ws;   // needs ~500 KB of scratch

  // repack weights to bf16 fragment layout every call (deterministic, cheap)
  {
    int total;
    total = (64 >> 5) * (64 >> 4) * 32;
    pack_kernel<<<(total + 127) / 128, 128, 0, stream>>>(rad_w1, pack + RW1_OFF, 64, 64);
    total = (64 >> 5) * (768 >> 4) * 32;
    pack_kernel<<<(total + 127) / 128, 128, 0, stream>>>(rad_w2, pack + RW2_OFF, 64, 768);
    total = (384 >> 5) * (384 >> 4) * 32;
    pack_kernel<<<(total + 127) / 128, 128, 0, stream>>>(lin_w0, pack + LW0_OFF, 384, 384);
    total = (384 >> 5) * (128 >> 4) * 32;
    pack_kernel<<<(total + 127) / 128, 128, 0, stream>>>(lin_w1, pack + LW1_OFF, 384, 128);
  }

  // dynamic LDS: 124.5 KB/WG -> 2 workgroups per 320 KB WGP
  static_assert(SMEM_BYTES <= 160 * 1024, "LDS budget");
  hipFuncSetAttribute((const void*)fctp_main,
                      hipFuncAttributeMaxDynamicSharedMemorySize, SMEM_BYTES);

  fctp_main<<<EDGES / EPB, 128, SMEM_BYTES, stream>>>(
      node_input, edge_attr, edge_scalars, rad_ln_g, rad_ln_b, rad_offset,
      lin_b0, pack, out);
}
